// GravityBrain_79774722556519
// MI455X (gfx1250) — matrix operand order repaired and checked by reading, not compiled, not run on hardware
//
#include <hip/hip_runtime.h>
#include <cstdint>

// ---------------------------------------------------------------------------
// Types
// ---------------------------------------------------------------------------
typedef __attribute__((ext_vector_type(16))) __bf16 v16bf;
typedef __attribute__((ext_vector_type(8)))  float  v8f;
typedef __attribute__((ext_vector_type(4)))  float  fvec4;
typedef __attribute__((ext_vector_type(2)))  float  fvec2;

#define D_     1024
#define H_     16
#define HD_    64
#define L_     8
#define T_     1024
#define B_     2
#define MROWS  (B_ * T_)   // 2048
#define HID    2734
#define HIDP   2736        // padded leading dim (16B-aligned rows)
#define VOCAB_ 32000

union Frag16 { v16bf v; fvec4 f4[2]; };

__device__ __forceinline__ float  bf2f(__bf16 b) { return (float)b; }
__device__ __forceinline__ __bf16 f2bf(float f)  { return (__bf16)f; }

// ---------------------------------------------------------------------------
// CDNA5 async global->LDS copy (ASYNCcnt path). GV mode: per-lane 64-bit
// global address + per-lane LDS byte offset (low 32 bits of flat shared ptr).
// ---------------------------------------------------------------------------
__device__ __forceinline__ void async_copy_b128(void* lds_dst, const void* gsrc) {
  uint32_t lds_off = (uint32_t)(uintptr_t)lds_dst;
  asm volatile("global_load_async_to_lds_b128 %0, %1, off"
               :: "v"(lds_off), "v"(gsrc) : "memory");
}
__device__ __forceinline__ void wait_async0() {
  asm volatile("s_wait_asynccnt 0" ::: "memory");
}

// ---------------------------------------------------------------------------
// Embedding gather: x[row,:] = W_emb[idx[row],:]
// ---------------------------------------------------------------------------
__global__ __launch_bounds__(256) void embed_kernel(const int* __restrict__ idx,
                                                    const float* __restrict__ emb,
                                                    float* __restrict__ x) {
  int row = blockIdx.x, tid = threadIdx.x;
  int tok = idx[row];
  fvec4 v = *(const fvec4*)(emb + (size_t)tok * D_ + tid * 4);
  *(fvec4*)(x + (size_t)row * D_ + tid * 4) = v;
}

// ---------------------------------------------------------------------------
// RMSNorm: out(bf16) = x * rsqrt(mean(x^2)+eps) * w
// ---------------------------------------------------------------------------
__global__ __launch_bounds__(256) void rmsnorm_kernel(const float* __restrict__ x,
                                                      const float* __restrict__ w,
                                                      __bf16* __restrict__ out) {
  int row = blockIdx.x, tid = threadIdx.x;
  fvec4 v = *(const fvec4*)(x + (size_t)row * D_ + tid * 4);
  float ss = v.x * v.x + v.y * v.y + v.z * v.z + v.w * v.w;
#pragma unroll
  for (int off = 16; off > 0; off >>= 1) ss += __shfl_xor(ss, off, 32);
  __shared__ float red[8];
  if ((tid & 31) == 0) red[tid >> 5] = ss;
  __syncthreads();
  float tot = 0.f;
#pragma unroll
  for (int i = 0; i < 8; ++i) tot += red[i];
  float inv = rsqrtf(tot * (1.0f / (float)D_) + 1e-6f);
  fvec4 wv = *(const fvec4*)(w + tid * 4);
  __bf16* o = out + (size_t)row * D_ + tid * 4;
  o[0] = f2bf(v.x * inv * wv.x);
  o[1] = f2bf(v.y * inv * wv.y);
  o[2] = f2bf(v.z * inv * wv.z);
  o[3] = f2bf(v.w * inv * wv.w);
}

// ---------------------------------------------------------------------------
// RoPE in-place on bf16 qkv rows [row, 3*D]: rotate q and k per head.
// block = 512 threads: h = tid>>5 (16 heads), i = tid&31 (half of HD)
// ---------------------------------------------------------------------------
__global__ __launch_bounds__(512) void rope_kernel(__bf16* __restrict__ qkv) {
  int row = blockIdx.x;
  int t = row & (T_ - 1);
  int h = threadIdx.x >> 5;
  int i = threadIdx.x & 31;
  float inv_freq = powf(10000.0f, -(float)(2 * i) / (float)HD_);
  float ang = (float)t * inv_freq;
  float c = cosf(ang), s = sinf(ang);
  size_t base = (size_t)row * (3 * D_) + (size_t)h * HD_;
#pragma unroll
  for (int qk = 0; qk < 2; ++qk) {
    __bf16* p = qkv + base + qk * D_;
    float x1 = bf2f(p[i]), x2 = bf2f(p[i + 32]);
    p[i]      = f2bf(x1 * c - x2 * s);
    p[i + 32] = f2bf(x1 * s + x2 * c);
  }
}

// ---------------------------------------------------------------------------
// SiLU-mul: g = silu(g) * u  (bf16 in/out, f32 math), covers padded width.
// ---------------------------------------------------------------------------
__global__ __launch_bounds__(256) void silumul_kernel(__bf16* __restrict__ g,
                                                      const __bf16* __restrict__ u,
                                                      int total) {
  int i = blockIdx.x * 256 + threadIdx.x;
  if (i >= total) return;
  float gf = bf2f(g[i]), uf = bf2f(u[i]);
  float sv = gf / (1.0f + __expf(-gf));
  g[i] = f2bf(sv * uf);
}

// ---------------------------------------------------------------------------
// WMMA GEMM: C[M,N] = A(bf16)[M,K] * B(f32->bf16)[K,N]
//   TRANSB=false: B[k][n] = Bsrc[k*ldb + n]   (weight stored KxN)
//   TRANSB=true : B[k][n] = Bsrc[n*ldb + k]   (weight stored NxK, e.g. W_emb)
//   EPI: 0 = store f32, 1 = store bf16 (zero-fill cols [N,ldc)), 2 = f32 +=
// Tile 128x128x32, 256 threads (8 waves), wave tile 64x32 (4x2 WMMA frags).
// A tile staging uses CDNA5 async global->LDS (bf16 data, no conversion);
// B tile staging converts f32->bf16 through VGPRs (weights are L2-resident).
// ---------------------------------------------------------------------------
template <int EPI, bool TRANSB>
__global__ __launch_bounds__(256) void gemm_wmma(const __bf16* __restrict__ A, int lda,
                                                 const float* __restrict__ Bsrc, int ldb,
                                                 void* __restrict__ Cp, int ldc,
                                                 int M, int N, int K) {
  __shared__ __align__(16) __bf16 As[128][40];   // [m][k], padded
  __shared__ __align__(16) __bf16 Bs[128][40];   // [n][k], padded (transposed tile)

  const int tid  = threadIdx.x;
  const int lane = tid & 31;
  const int wave = tid >> 5;
  const int wm   = (wave & 1) * 64;
  const int wn   = (wave >> 1) * 32;
  const int row0 = blockIdx.y * 128;
  const int col0 = blockIdx.x * 128;
  const int hi   = lane >> 4;
  const int ln   = lane & 15;

  v8f acc[4][2] = {};
  const int ktiles = (K + 31) >> 5;

  for (int kt = 0; kt < ktiles; ++kt) {
    const int k0 = kt << 5;
    __syncthreads();

    // ---- stage A tile [128][32] (bf16 global -> LDS, async path) ----
#pragma unroll
    for (int it = 0; it < 2; ++it) {
      int chunk = tid + it * 256;
      int r = chunk >> 2, c = (chunk & 3) << 3;
      int gk = k0 + c;
      const __bf16* src = A + (size_t)(row0 + r) * lda + gk;
      if (gk + 8 <= lda) {
        async_copy_b128(&As[r][c], src);
      } else {
        __align__(16) __bf16 tmp[8];
#pragma unroll
        for (int j = 0; j < 8; ++j) tmp[j] = (gk + j < lda) ? src[j] : f2bf(0.f);
        *(fvec4*)(&As[r][c]) = *(const fvec4*)tmp;
      }
    }

    // ---- stage B tile -> Bs[n][k] with f32->bf16 convert ----
    if (TRANSB) {
#pragma unroll
      for (int it = 0; it < 4; ++it) {
        int chunk = tid + it * 256;
        int n = chunk >> 3, kc = (chunk & 7) << 2;
        int gn = col0 + n, gk = k0 + kc;
        fvec4 v = {0.f, 0.f, 0.f, 0.f};
        if (gn < N) {
          const float* src = Bsrc + (size_t)gn * ldb + gk;
          if (gk + 4 <= K) {
            v = *(const fvec4*)src;
          } else {
#pragma unroll
            for (int j = 0; j < 4; ++j) v[j] = (gk + j < K) ? src[j] : 0.f;
          }
        }
        __align__(8) __bf16 b4[4] = { f2bf(v.x), f2bf(v.y), f2bf(v.z), f2bf(v.w) };
        *(fvec2*)(&Bs[n][kc]) = *(const fvec2*)b4;
      }
    } else {
#pragma unroll
      for (int it = 0; it < 8; ++it) {
        int chunk = tid + it * 256;
        int kk = chunk >> 6, n2 = (chunk & 63) << 1;
        int gk = k0 + kk, gn = col0 + n2;
        fvec2 v = {0.f, 0.f};
        if (gk < K) {
          const float* src = Bsrc + (size_t)gk * ldb + gn;
          if (gn + 2 <= N) v = *(const fvec2*)src;
          else if (gn < N) v.x = src[0];
        }
        Bs[n2][kk]     = f2bf(v.x);
        Bs[n2 + 1][kk] = f2bf(v.y);
      }
    }
    wait_async0();          // A-tile async copies done (per issuing thread)
    __syncthreads();        // publish As/Bs block-wide

    // prefetch next B tile region into GL2 (speculative)
    if (kt + 1 < ktiles && lane == 0) {
      const float* nb = TRANSB ? (Bsrc + (size_t)col0 * ldb + (k0 + 32))
                               : (Bsrc + (size_t)(k0 + 32) * ldb + col0);
      __builtin_prefetch(nb, 0, 0);
    }

    // ---- fragments + WMMA ----
    Frag16 bfr[2];
#pragma unroll
    for (int j = 0; j < 2; ++j) {
      int n  = wn + j * 16 + ln;
      int ks = hi << 4;                       // lanes 0-15: K 0..15, 16-31: K 16..31
      bfr[j].f4[0] = *(const fvec4*)(&Bs[n][ks]);
      bfr[j].f4[1] = *(const fvec4*)(&Bs[n][ks + 8]);
    }
#pragma unroll
    for (int i = 0; i < 4; ++i) {
      Frag16 af;
      int m  = wm + i * 16 + ln;
      int ks = hi << 4;                       // lanes 0-15: K 0..15, 16-31: K 16..31
      af.f4[0] = *(const fvec4*)(&As[m][ks]);
      af.f4[1] = *(const fvec4*)(&As[m][ks + 8]);
#pragma unroll
      for (int j = 0; j < 2; ++j)
        acc[i][j] = __builtin_amdgcn_wmma_f32_16x16x32_bf16(
            false, af.v, false, bfr[j].v, (short)0, acc[i][j], false, false);
    }
  }

  // ---- epilogue (C layout: VGPR r -> M = r + 8*(lane>=16), N = lane%16) ----
#pragma unroll
  for (int i = 0; i < 4; ++i)
#pragma unroll
    for (int j = 0; j < 2; ++j)
#pragma unroll
      for (int r = 0; r < 8; ++r) {
        int row = row0 + wm + i * 16 + r + hi * 8;
        int col = col0 + wn + j * 16 + ln;
        float v = acc[i][j][r];
        if (EPI == 0) {
          if (col < N) ((float*)Cp)[(size_t)row * ldc + col] = v;
        } else if (EPI == 1) {
          __bf16* C = (__bf16*)Cp;
          if (col < N)        C[(size_t)row * ldc + col] = f2bf(v);
          else if (col < ldc) C[(size_t)row * ldc + col] = f2bf(0.f);
        } else {
          if (col < N) {
            float* C = (float*)Cp;
            size_t o = (size_t)row * ldc + col;
            C[o] = C[o] + v;
          }
        }
      }
}

// ---------------------------------------------------------------------------
// Flash attention: grid (T/128, B*H), 256 threads = 8 waves, wave = 16 Q rows.
// qkv rows: [q(H,64) | k(H,64) | v(H,64)] bf16. Output o[row, h*64+hd] bf16.
// Q/K tiles staged via async global->LDS; V staged through VGPRs (transpose).
// ---------------------------------------------------------------------------
__global__ __launch_bounds__(256) void attn_kernel(const __bf16* __restrict__ qkv,
                                                   __bf16* __restrict__ o_out) {
  __shared__ __align__(16) __bf16 Qs[128][72];       // [q][hd]
  __shared__ __align__(16) __bf16 Ks[128][72];       // [key][hd]
  __shared__ __align__(16) __bf16 Vt[64][136];       // [hd][key] (transposed)
  __shared__ __align__(16) __bf16 Ps[8][16][128];    // per-wave P tile

  const int tid = threadIdx.x, lane = tid & 31, wave = tid >> 5;
  const int qt = blockIdx.x;                  // q tile 0..7
  const int bh = blockIdx.y;                  // 0..31
  const int b = bh >> 4, h = bh & 15;
  const size_t brow = (size_t)b * T_;
  const int hi = lane >> 4, ln = lane & 15;

  // stage Q tile once (async)
#pragma unroll
  for (int it = 0; it < 4; ++it) {
    int chunk = tid + it * 256;
    int r = chunk >> 3, c = (chunk & 7) << 3;
    size_t grow = brow + (size_t)qt * 128 + r;
    async_copy_b128(&Qs[r][c], qkv + grow * 3072 + h * 64 + c);
  }
  wait_async0();

  v8f oacc[4] = {};
  float mrow[8], lrow[8];
#pragma unroll
  for (int r = 0; r < 8; ++r) { mrow[r] = -1e30f; lrow[r] = 0.f; }

  for (int kvt = 0; kvt <= qt; ++kvt) {
    __syncthreads();
    // stage K tile (async) and transposed V tile (VALU)
#pragma unroll
    for (int it = 0; it < 4; ++it) {
      int chunk = tid + it * 256;
      int r = chunk >> 3, c = (chunk & 7) << 3;
      size_t grow = brow + (size_t)kvt * 128 + r;
      async_copy_b128(&Ks[r][c], qkv + grow * 3072 + 1024 + h * 64 + c);
      union { fvec4 f; __bf16 e[8]; } vv;
      vv.f = *(const fvec4*)(qkv + grow * 3072 + 2048 + h * 64 + c);
#pragma unroll
      for (int j = 0; j < 8; ++j) Vt[c + j][r] = vv.e[j];
    }
    wait_async0();
    __syncthreads();

    // S = Q @ K^T  (wave's 16 rows x 128 keys), contraction HD=64 = 2 steps
    v8f s[8] = {};
#pragma unroll
    for (int kk = 0; kk < 2; ++kk) {
      Frag16 qa;
      int m  = wave * 16 + ln;
      int ks = kk * 32 + (hi << 4);
      qa.f4[0] = *(const fvec4*)(&Qs[m][ks]);
      qa.f4[1] = *(const fvec4*)(&Qs[m][ks + 8]);
#pragma unroll
      for (int j = 0; j < 8; ++j) {
        Frag16 kb;
        int n   = j * 16 + ln;
        int ks2 = kk * 32 + (hi << 4);
        kb.f4[0] = *(const fvec4*)(&Ks[n][ks2]);
        kb.f4[1] = *(const fvec4*)(&Ks[n][ks2 + 8]);
        s[j] = __builtin_amdgcn_wmma_f32_16x16x32_bf16(
            false, qa.v, false, kb.v, (short)0, s[j], false, false);
      }
    }

    // online softmax (rows live in lane halves; reduce across 16 lanes)
    float alpha[8];
#pragma unroll
    for (int r = 0; r < 8; ++r) {
      int grow = qt * 128 + wave * 16 + r + hi * 8;
      float mx = -1e30f;
#pragma unroll
      for (int j = 0; j < 8; ++j) {
        int gcol = kvt * 128 + j * 16 + ln;
        float v = s[j][r] * 0.125f;           // 1/sqrt(64)
        if (gcol > grow) v = -1e30f;          // causal mask
        s[j][r] = v;
        mx = fmaxf(mx, v);
      }
#pragma unroll
      for (int off = 1; off < 16; off <<= 1) mx = fmaxf(mx, __shfl_xor(mx, off, 16));
      float mnew = fmaxf(mrow[r], mx);
      alpha[r] = __expf(mrow[r] - mnew);
      float sum = 0.f;
#pragma unroll
      for (int j = 0; j < 8; ++j) {
        float p = __expf(s[j][r] - mnew);
        Ps[wave][r + hi * 8][j * 16 + ln] = f2bf(p);
        sum += p;
      }
#pragma unroll
      for (int off = 1; off < 16; off <<= 1) sum += __shfl_xor(sum, off, 16);
      lrow[r] = lrow[r] * alpha[r] + sum;
      mrow[r] = mnew;
    }
#pragma unroll
    for (int jo = 0; jo < 4; ++jo)
#pragma unroll
      for (int r = 0; r < 8; ++r) oacc[jo][r] *= alpha[r];

    // intra-wave LDS RAW: P stores -> P fragment loads
    asm volatile("s_wait_dscnt 0" ::: "memory");

    // O += P @ V  (contraction 128 keys = 4 steps)
#pragma unroll
    for (int kk = 0; kk < 4; ++kk) {
      Frag16 pa;
      int ks = kk * 32 + (hi << 4);
      pa.f4[0] = *(const fvec4*)(&Ps[wave][ln][ks]);
      pa.f4[1] = *(const fvec4*)(&Ps[wave][ln][ks + 8]);
#pragma unroll
      for (int jo = 0; jo < 4; ++jo) {
        Frag16 vb;
        int n   = jo * 16 + ln;
        int ks2 = kk * 32 + (hi << 4);
        vb.f4[0] = *(const fvec4*)(&Vt[n][ks2]);
        vb.f4[1] = *(const fvec4*)(&Vt[n][ks2 + 8]);
        oacc[jo] = __builtin_amdgcn_wmma_f32_16x16x32_bf16(
            false, pa.v, false, vb.v, (short)0, oacc[jo], false, false);
      }
    }
  }

  // normalize + write o[row, h*64 + hd]
#pragma unroll
  for (int jo = 0; jo < 4; ++jo)
#pragma unroll
    for (int r = 0; r < 8; ++r) {
      size_t grow = brow + (size_t)qt * 128 + wave * 16 + r + hi * 8;
      int gcol = h * 64 + jo * 16 + ln;
      o_out[grow * 1024 + gcol] = f2bf(oacc[jo][r] / lrow[r]);
    }
}

// ---------------------------------------------------------------------------
// Host orchestration
// ---------------------------------------------------------------------------
extern "C" void kernel_launch(void* const* d_in, const int* in_sizes, int n_in,
                              void* d_out, int out_size, void* d_ws, size_t ws_size,
                              hipStream_t stream) {
  (void)in_sizes; (void)n_in; (void)out_size; (void)ws_size;
  const int*   idx   = (const int*)d_in[0];
  const float* W_emb = (const float*)d_in[1];
  const float* norm1 = (const float*)d_in[2];
  const float* Wqkv  = (const float*)d_in[3];
  const float* Wout  = (const float*)d_in[4];
  const float* norm2 = (const float*)d_in[5];
  const float* Wg    = (const float*)d_in[6];
  const float* Wu    = (const float*)d_in[7];
  const float* Wd    = (const float*)d_in[8];
  const float* fnorm = (const float*)d_in[9];
  float* out = (float*)d_out;

  char* ws = (char*)d_ws;
  size_t off = 0;
  auto take = [&](size_t bytes) {
    char* p = ws + off;
    off = (off + bytes + 255) & ~(size_t)255;
    return p;
  };
  float*  x    = (float*) take((size_t)MROWS * D_ * 4);       // 8 MB residual stream
  __bf16* hbuf = (__bf16*)take((size_t)MROWS * D_ * 2);       // 4 MB normed acts
  __bf16* qkv  = (__bf16*)take((size_t)MROWS * 3 * D_ * 2);   // 12.6 MB
  __bf16* obuf = (__bf16*)take((size_t)MROWS * D_ * 2);       // 4 MB attn out
  __bf16* g    = (__bf16*)take((size_t)MROWS * HIDP * 2);     // 11.2 MB
  __bf16* u    = (__bf16*)take((size_t)MROWS * HIDP * 2);     // 11.2 MB

  embed_kernel<<<MROWS, 256, 0, stream>>>(idx, W_emb, x);

  for (int l = 0; l < L_; ++l) {
    rmsnorm_kernel<<<MROWS, 256, 0, stream>>>(x, norm1 + (size_t)l * D_, hbuf);
    gemm_wmma<1, false><<<dim3(24, 16), 256, 0, stream>>>(
        hbuf, D_, Wqkv + (size_t)l * D_ * 3 * D_, 3 * D_, qkv, 3 * D_,
        MROWS, 3 * D_, D_);
    rope_kernel<<<MROWS, 512, 0, stream>>>(qkv);
    attn_kernel<<<dim3(T_ / 128, B_ * H_), 256, 0, stream>>>(qkv, obuf);
    gemm_wmma<2, false><<<dim3(8, 16), 256, 0, stream>>>(
        obuf, D_, Wout + (size_t)l * D_ * D_, D_, x, D_, MROWS, D_, D_);
    rmsnorm_kernel<<<MROWS, 256, 0, stream>>>(x, norm2 + (size_t)l * D_, hbuf);
    gemm_wmma<1, false><<<dim3(22, 16), 256, 0, stream>>>(
        hbuf, D_, Wg + (size_t)l * D_ * HID, HID, g, HIDP, MROWS, HID, D_);
    gemm_wmma<1, false><<<dim3(22, 16), 256, 0, stream>>>(
        hbuf, D_, Wu + (size_t)l * D_ * HID, HID, u, HIDP, MROWS, HID, D_);
    int total = MROWS * HIDP;
    silumul_kernel<<<(total + 255) / 256, 256, 0, stream>>>(g, u, total);
    gemm_wmma<2, false><<<dim3(8, 16), 256, 0, stream>>>(
        g, HIDP, Wd + (size_t)l * HID * D_, D_, x, D_, MROWS, D_, HID);
  }

  rmsnorm_kernel<<<MROWS, 256, 0, stream>>>(x, fnorm, hbuf);
  // logits = h @ W_emb^T  (W_emb is [VOCAB, D] -> TRANSB)
  gemm_wmma<0, true><<<dim3(VOCAB_ / 128, 16), 256, 0, stream>>>(
      hbuf, D_, W_emb, D_, out, VOCAB_, MROWS, VOCAB_, D_);
}
